// TransformerEncoderLayer_4999341933084
// MI455X (gfx1250) — compile-verified, matrix operand order checked by reading, not run on hardware
//
#include <hip/hip_runtime.h>

typedef __attribute__((ext_vector_type(16))) __bf16 v16bf;
typedef __attribute__((ext_vector_type(8)))  float  v8f;
typedef __attribute__((ext_vector_type(4)))  float  f32x4;
typedef __attribute__((ext_vector_type(4)))  unsigned int u32x4;
typedef __attribute__((ext_vector_type(8)))  int    i32x8;
typedef __attribute__((ext_vector_type(4)))  int    i32x4;

#define NBL    8192   // B*L
#define DMODEL 256
#define NHEAD  8
#define DH     2048   // D*H
#define LSEQ   2048
#define FFD    128

// ---------------------------------------------------------------- helpers
__device__ __forceinline__ __bf16 to_bf16(float f) {
  union { float f; unsigned u; } x; x.f = f;
  unsigned r = x.u + 0x7FFFu + ((x.u >> 16) & 1u);   // round-to-nearest-even
  unsigned short h = (unsigned short)(r >> 16);
  union { unsigned short s; __bf16 b; } y; y.s = h;
  return y.b;
}

// A/B fragment for v_wmma_f32_16x16x32_bf16 from a row-major (rows x ld) bf16
// matrix (global or LDS), contraction chunk starting at column kc.
// Per CDNA5 ISA 16-bit layout: lane holds row (lane&15); elements 0..7 =
// K [off, off+8), elements 8..15 = K [off+16, off+24), off = (lane&16)?8:0.
__device__ __forceinline__ v16bf load_frag(const __bf16* base, int ld, int row,
                                           int kc, int lane) {
  const __bf16* p = base + (size_t)row * ld + kc + ((lane & 16) ? 8 : 0);
  v16bf f;
  ((f32x4*)&f)[0] = ((const f32x4*)p)[0];
  ((f32x4*)&f)[1] = *(const f32x4*)(p + 16);
  return f;
}

__device__ __forceinline__ v8f wmma_bf16(v16bf a, v16bf b, v8f c) {
  return __builtin_amdgcn_wmma_f32_16x16x32_bf16(false, a, false, b,
                                                 (short)0, c, false, false);
}

__device__ __forceinline__ unsigned lds_off(const void* p) {
  // generic pointer to a __shared__ object: low 32 bits = LDS byte offset
  return (unsigned)(unsigned long long)p;
}

// Tensor Data Mover: 2D bf16 tile (tile0 contiguous elems x tile1 rows,
// row stride = stride0 elems) from global -> LDS, packed rows in LDS.
// Tracked by TENSORcnt.
__device__ __forceinline__ void tdm_load_2d(unsigned lds_addr, const void* gptr,
                                            unsigned tile0, unsigned tile1,
                                            unsigned stride0) {
  unsigned long long ga = (unsigned long long)gptr;
  u32x4 g0;
  g0[0] = 1u;                                        // count=1, user descriptor
  g0[1] = lds_addr;                                  // lds_addr
  g0[2] = (unsigned)(ga & 0xFFFFFFFFu);              // global_addr[31:0]
  g0[3] = (unsigned)((ga >> 32) & 0x01FFFFFFu)       // global_addr[56:32]
          | (2u << 30);                              // type = 2 ("image")
  i32x8 g1;
  g1[0] = (int)(1u << 16);          // workgroup_mask=0, data_size=1 (2 bytes)
  g1[1] = (int)(tile0 << 16);       // tensor_dim0[15:0]  (= tile -> no OOB)
  g1[2] = (int)(tile1 << 16);       // tensor_dim0[31:16]=0, tensor_dim1[15:0]
  g1[3] = (int)(tile0 << 16);       // tensor_dim1[31:16]=0, tile_dim0
  g1[4] = (int)(tile1 & 0xFFFFu);   // tile_dim1, tile_dim2=0
  g1[5] = (int)stride0;             // tensor_dim0_stride[31:0] (elements)
  g1[6] = 0;                        // stride0[47:32]=0, dim1_stride[15:0]=0
  g1[7] = 0;
  i32x4 z4 = {0, 0, 0, 0};
#if defined(__clang_major__) && (__clang_major__ >= 23)
  i32x8 z8 = {0, 0, 0, 0, 0, 0, 0, 0};
  __builtin_amdgcn_tensor_load_to_lds(g0, g1, z4, z4, z8, 0);
#else
  __builtin_amdgcn_tensor_load_to_lds(g0, g1, z4, z4, 0);
#endif
}

// ---------------------------------------------------------------- converts
__global__ void f32_to_bf16_kernel(const float* __restrict__ src,
                                   __bf16* __restrict__ dst, int n) {
  int i = blockIdx.x * blockDim.x + threadIdx.x;
  if (i < n) dst[i] = to_bf16(src[i]);
}

// src (rows x cols) f32  ->  dst (cols x rows) bf16   (weights -> NxK form)
__global__ void transpose_bf16_kernel(const float* __restrict__ src,
                                      __bf16* __restrict__ dst,
                                      int rows, int cols) {
  int i = blockIdx.x * blockDim.x + threadIdx.x;
  if (i >= rows * cols) return;
  int rr = i / cols, cc = i - rr * cols;
  dst[(size_t)cc * rows + rr] = to_bf16(src[i]);
}

// ---------------------------------------------------------------- GEMM
// C(M,N) = A(M,K) x Bt(N,K)^T, bf16 in, f32 accumulate.  The 64x32 Bt tile is
// shared by all 4 waves: streamed into LDS by the TDM, double-buffered.
// OUTMODE: 0 = f32 row-major, 1 = bf16 row-major, 2 = bf16 transposed (ldT)
template <int OUTMODE, bool RELU>
__global__ void __launch_bounds__(128, 2)
gemm_bf16_kernel(const __bf16* __restrict__ A, const __bf16* __restrict__ Bt,
                 const float* __restrict__ bias, float* __restrict__ outF,
                 __bf16* __restrict__ outB, int M, int N, int K, int ldT) {
  __shared__ __bf16 btile[2][64 * 32];   // (n, k) rows packed, 4 KB each
  const int lane = threadIdx.x & 31;
  const int wave = threadIdx.x >> 5;
  const int m0 = blockIdx.y * 64 + wave * 16;
  const int n0 = blockIdx.x * 64;
  const int niter = K >> 5;

  if (wave == 0)
    tdm_load_2d(lds_off(&btile[0][0]), Bt + (size_t)n0 * K, 32, 64, (unsigned)K);

  v8f acc0 = {}, acc1 = {}, acc2 = {}, acc3 = {};
  const int arow = m0 + (lane & 15);
  for (int i = 0; i < niter; ++i) {
    if (wave == 0) {
      if (i + 1 < niter) {
        tdm_load_2d(lds_off(&btile[(i + 1) & 1][0]),
                    Bt + (size_t)n0 * K + (i + 1) * 32, 32, 64, (unsigned)K);
        __builtin_amdgcn_s_wait_tensorcnt(1);   // current tile landed
      } else {
        __builtin_amdgcn_s_wait_tensorcnt(0);
      }
    }
    __syncthreads();
    const __bf16* bt = btile[i & 1];
    v16bf a  = load_frag(A, K, arow, i * 32, lane);
    v16bf b0 = load_frag(bt, 32, 0 * 16 + (lane & 15), 0, lane);
    v16bf b1 = load_frag(bt, 32, 1 * 16 + (lane & 15), 0, lane);
    v16bf b2 = load_frag(bt, 32, 2 * 16 + (lane & 15), 0, lane);
    v16bf b3 = load_frag(bt, 32, 3 * 16 + (lane & 15), 0, lane);
    acc0 = wmma_bf16(a, b0, acc0);
    acc1 = wmma_bf16(a, b1, acc1);
    acc2 = wmma_bf16(a, b2, acc2);
    acc3 = wmma_bf16(a, b3, acc3);
    __syncthreads();   // tile consumed before it is overwritten
  }

  v8f accs[4] = {acc0, acc1, acc2, acc3};
#pragma unroll
  for (int nt = 0; nt < 4; ++nt) {
#pragma unroll
    for (int r = 0; r < 8; ++r) {
      int Mi = m0 + r + ((lane & 16) ? 8 : 0);
      int Ni = n0 + nt * 16 + (lane & 15);
      float v = accs[nt][r];
      if (bias) v += bias[Ni];
      if (RELU) v = fmaxf(v, 0.0f);
      if (OUTMODE == 0)      outF[(size_t)Mi * N + Ni] = v;
      else if (OUTMODE == 1) outB[(size_t)Mi * N + Ni] = to_bf16(v);
      else                   outB[(size_t)Ni * ldT + Mi] = to_bf16(v);
    }
  }
}

// ---------------------------------------------------------------- flash attn
// Block = 4 waves sharing one (b,h); each wave owns a 16-row q-tile.  K and V
// 32-key tiles are TDM-streamed into LDS (double-buffered) and shared by all
// waves.  q,k: (NBL, DH) bf16; vT: (DH, NBL) bf16; heads: (NBL, DH) bf16.
__global__ void __launch_bounds__(128, 1)
flash_attn_kernel(const __bf16* __restrict__ q, const __bf16* __restrict__ k,
                  const __bf16* __restrict__ vT, __bf16* __restrict__ heads) {
  __shared__ __bf16 ktile[2][32 * 256];   // (key, d)   16 KB each
  __shared__ __bf16 vtile[2][256 * 32];   // (d, key)   16 KB each
  __shared__ __bf16 psh[4][16 * 32];      // per-wave P relayout, 1 KB each
  const int lane = threadIdx.x & 31;
  const int wave = threadIdx.x >> 5;
  const int w = blockIdx.x * 4 + wave;    // 0 .. 4095
  const int b  = w >> 10;                 // / (H * L/16)
  const int h  = (w >> 7) & 7;            // same h for all 4 waves of a block
  const int qt = w & 127;
  __bf16* ps = psh[wave];
  const int NT = LSEQ / 32;

  const __bf16* kbase = k  + (size_t)(b * LSEQ) * DH + h * DMODEL;
  const __bf16* vbase = vT + (size_t)(h * DMODEL) * NBL + b * LSEQ;

  if (wave == 0) {   // prefetch tile 0
    tdm_load_2d(lds_off(&ktile[0][0]), kbase, 256, 32, DH);
    tdm_load_2d(lds_off(&vtile[0][0]), vbase, 32, 256, NBL);
  }

  // Q fragments: 8 K-chunks covering d = 0..255 (global, loaded once)
  v16bf qa[8];
  const int qrow = b * LSEQ + qt * 16 + (lane & 15);
#pragma unroll
  for (int c = 0; c < 8; ++c)
    qa[c] = load_frag(q, DH, qrow, h * DMODEL + c * 32, lane);

  v8f o[16];
  v8f zero = {};
#pragma unroll
  for (int nt = 0; nt < 16; ++nt) o[nt] = zero;
  float mrow[8], lrow[8];
#pragma unroll
  for (int r = 0; r < 8; ++r) { mrow[r] = -1e30f; lrow[r] = 0.0f; }

  const float scale = 0.0625f;  // 1/sqrt(256)
  for (int t = 0; t < NT; ++t) {
    if (wave == 0) {
      if (t + 1 < NT) {   // stream next K/V pair while computing on current
        tdm_load_2d(lds_off(&ktile[(t + 1) & 1][0]),
                    kbase + (size_t)((t + 1) * 32) * DH, 256, 32, DH);
        tdm_load_2d(lds_off(&vtile[(t + 1) & 1][0]),
                    vbase + (t + 1) * 32, 32, 256, NBL);
        __builtin_amdgcn_s_wait_tensorcnt(2);   // current pair landed
      } else {
        __builtin_amdgcn_s_wait_tensorcnt(0);
      }
    }
    __syncthreads();
    const __bf16* kt = ktile[t & 1];
    const __bf16* vt = vtile[t & 1];

    // --- S = Q * K^T  (16 x 32 scores, two 16x16 WMMA tiles per K-chunk)
    v8f s0 = zero, s1 = zero;
#pragma unroll
    for (int c = 0; c < 8; ++c) {
      v16bf kb0 = load_frag(kt, 256, (lane & 15),      c * 32, lane);
      v16bf kb1 = load_frag(kt, 256, 16 + (lane & 15), c * 32, lane);
      s0 = wmma_bf16(qa[c], kb0, s0);
      s1 = wmma_bf16(qa[c], kb1, s1);
    }

    // --- online softmax (row M = r + (lane>=16)*8 lives in one 16-lane half)
#pragma unroll
    for (int r = 0; r < 8; ++r) {
      float a0 = s0[r] * scale, a1 = s1[r] * scale;
      float mx = fmaxf(a0, a1);
      mx = fmaxf(mx, __shfl_xor(mx, 1));
      mx = fmaxf(mx, __shfl_xor(mx, 2));
      mx = fmaxf(mx, __shfl_xor(mx, 4));
      mx = fmaxf(mx, __shfl_xor(mx, 8));
      float mnew = fmaxf(mrow[r], mx);
      float corr = __expf(mrow[r] - mnew);
      float p0 = __expf(a0 - mnew);
      float p1 = __expf(a1 - mnew);
      float sum = p0 + p1;
      sum += __shfl_xor(sum, 1);
      sum += __shfl_xor(sum, 2);
      sum += __shfl_xor(sum, 4);
      sum += __shfl_xor(sum, 8);
      lrow[r] = lrow[r] * corr + sum;
      mrow[r] = mnew;
#pragma unroll
      for (int nt = 0; nt < 16; ++nt) o[nt][r] *= corr;
      // stash P (C-layout element -> row-major 16x32 bf16 in LDS)
      const int rowi = r + ((lane & 16) ? 8 : 0);
      ps[rowi * 32 + (lane & 15)]      = to_bf16(p0);
      ps[rowi * 32 + 16 + (lane & 15)] = to_bf16(p1);
    }
    asm volatile("s_wait_dscnt 0" ::: "memory");

    // reload P as an A-fragment (16x32, K = 32 keys)
    v16bf pa = load_frag(ps, 32, (lane & 15), 0, lane);

    // --- O += P * V   (16 output tiles across d = 256)
#pragma unroll
    for (int nt = 0; nt < 16; ++nt) {
      v16bf vb = load_frag(vt, 32, nt * 16 + (lane & 15), 0, lane);
      o[nt] = wmma_bf16(pa, vb, o[nt]);
    }
    __syncthreads();   // tiles consumed before TDM overwrites them
  }

  // normalize and store heads (bf16 row-major (NBL, DH))
#pragma unroll
  for (int r = 0; r < 8; ++r) {
    float inv = 1.0f / lrow[r];
#pragma unroll
    for (int nt = 0; nt < 16; ++nt) o[nt][r] *= inv;
  }
  const int orow0 = b * LSEQ + qt * 16;
#pragma unroll
  for (int nt = 0; nt < 16; ++nt) {
#pragma unroll
    for (int r = 0; r < 8; ++r) {
      int Mi = orow0 + r + ((lane & 16) ? 8 : 0);
      int Ni = h * DMODEL + nt * 16 + (lane & 15);
      heads[(size_t)Mi * DH + Ni] = to_bf16(o[nt][r]);
    }
  }
}

// ---------------------------------------------------------------- add + LN
// y = LN(a + rsd) * g + b ;  one 256-thread block per row of 256 elements
__global__ void add_ln_kernel(const float* __restrict__ a,
                              const float* __restrict__ rsd,
                              const float* __restrict__ g,
                              const float* __restrict__ bta,
                              float* __restrict__ outF,
                              __bf16* __restrict__ outB) {
  __shared__ float red[256];
  const int row = blockIdx.x, t = threadIdx.x;
  float v = a[(size_t)row * 256 + t] + rsd[(size_t)row * 256 + t];
  red[t] = v; __syncthreads();
  for (int s = 128; s > 0; s >>= 1) { if (t < s) red[t] += red[t + s]; __syncthreads(); }
  float mu = red[0] * (1.0f / 256.0f);
  __syncthreads();
  float d = v - mu;
  red[t] = d * d; __syncthreads();
  for (int s = 128; s > 0; s >>= 1) { if (t < s) red[t] += red[t + s]; __syncthreads(); }
  float var = red[0] * (1.0f / 256.0f);
  float y = d * rsqrtf(var + 1e-5f) * g[t] + bta[t];
  outF[(size_t)row * 256 + t] = y;
  if (outB) outB[(size_t)row * 256 + t] = to_bf16(y);
}

// ---------------------------------------------------------------- launcher
extern "C" void kernel_launch(void* const* d_in, const int* in_sizes, int n_in,
                              void* d_out, int out_size, void* d_ws,
                              size_t ws_size, hipStream_t stream) {
  (void)in_sizes; (void)n_in; (void)out_size; (void)ws_size;
  const float* seq  = (const float*)d_in[0];
  const float* Wq   = (const float*)d_in[1];
  const float* Wk   = (const float*)d_in[2];
  const float* Wv   = (const float*)d_in[3];
  const float* Wo   = (const float*)d_in[4];
  const float* ln1g = (const float*)d_in[5];
  const float* ln1b = (const float*)d_in[6];
  const float* W1   = (const float*)d_in[7];
  const float* b1   = (const float*)d_in[8];
  const float* W2   = (const float*)d_in[9];
  const float* b2   = (const float*)d_in[10];
  const float* ln2g = (const float*)d_in[11];
  const float* ln2b = (const float*)d_in[12];

  char* w = (char*)d_ws;
  size_t off = 0;
  auto take = [&](size_t bytes) {
    size_t o = off; off += (bytes + 255) & ~(size_t)255; return o;
  };
  __bf16* seqb  = (__bf16*)(w + take((size_t)NBL * DMODEL * 2));
  __bf16* WqT   = (__bf16*)(w + take((size_t)DH * DMODEL * 2));
  __bf16* WkT   = (__bf16*)(w + take((size_t)DH * DMODEL * 2));
  __bf16* WvT   = (__bf16*)(w + take((size_t)DH * DMODEL * 2));
  __bf16* WoT   = (__bf16*)(w + take((size_t)DMODEL * DH * 2));
  __bf16* W1T   = (__bf16*)(w + take((size_t)FFD * DMODEL * 2));
  __bf16* W2T   = (__bf16*)(w + take((size_t)DMODEL * FFD * 2));
  __bf16* qb    = (__bf16*)(w + take((size_t)NBL * DH * 2));
  __bf16* kb    = (__bf16*)(w + take((size_t)NBL * DH * 2));
  __bf16* vT    = (__bf16*)(w + take((size_t)DH * NBL * 2));
  __bf16* heads = (__bf16*)(w + take((size_t)NBL * DH * 2));
  float*  tmpF  = (float*)(w + take((size_t)NBL * DMODEL * 4));
  float*  xF    = (float*)(w + take((size_t)NBL * DMODEL * 4));
  __bf16* xB    = (__bf16*)(w + take((size_t)NBL * DMODEL * 2));
  __bf16* h1    = (__bf16*)(w + take((size_t)NBL * FFD * 2));

  // 1) convert / transpose inputs to bf16
  {
    int n = NBL * DMODEL;
    f32_to_bf16_kernel<<<(n + 255) / 256, 256, 0, stream>>>(seq, seqb, n);
  }
  {
    int n = DMODEL * DH;
    transpose_bf16_kernel<<<(n + 255) / 256, 256, 0, stream>>>(Wq, WqT, DMODEL, DH);
    transpose_bf16_kernel<<<(n + 255) / 256, 256, 0, stream>>>(Wk, WkT, DMODEL, DH);
    transpose_bf16_kernel<<<(n + 255) / 256, 256, 0, stream>>>(Wv, WvT, DMODEL, DH);
    transpose_bf16_kernel<<<(n + 255) / 256, 256, 0, stream>>>(Wo, WoT, DH, DMODEL);
  }
  {
    int n = DMODEL * FFD;
    transpose_bf16_kernel<<<(n + 255) / 256, 256, 0, stream>>>(W1, W1T, DMODEL, FFD);
    transpose_bf16_kernel<<<(n + 255) / 256, 256, 0, stream>>>(W2, W2T, FFD, DMODEL);
  }

  // 2) QKV projections (V stored transposed for the P*V WMMA path)
  dim3 gQKV(DH / 64, NBL / 64);
  gemm_bf16_kernel<1, false><<<gQKV, 128, 0, stream>>>(seqb, WqT, nullptr, nullptr, qb, NBL, DH, DMODEL, 0);
  gemm_bf16_kernel<1, false><<<gQKV, 128, 0, stream>>>(seqb, WkT, nullptr, nullptr, kb, NBL, DH, DMODEL, 0);
  gemm_bf16_kernel<2, false><<<gQKV, 128, 0, stream>>>(seqb, WvT, nullptr, nullptr, vT, NBL, DH, DMODEL, NBL);

  // 3) fused flash attention -> heads
  flash_attn_kernel<<<(NBL / 16) * NHEAD / 4, 128, 0, stream>>>(qb, kb, vT, heads);

  // 4) output projection: att = heads @ Wo (f32)
  dim3 gO(DMODEL / 64, NBL / 64);
  gemm_bf16_kernel<0, false><<<gO, 128, 0, stream>>>(heads, WoT, nullptr, tmpF, nullptr, NBL, DMODEL, DH, 0);

  // 5) x = LN(seq + att)   (also bf16 copy for the FF GEMM)
  add_ln_kernel<<<NBL, 256, 0, stream>>>(tmpF, seq, ln1g, ln1b, xF, xB);

  // 6) FF: h1 = relu(x @ W1 + b1); ff = h1 @ W2 + b2
  dim3 gF1(FFD / 64, NBL / 64);
  gemm_bf16_kernel<1, true><<<gF1, 128, 0, stream>>>(xB, W1T, b1, nullptr, h1, NBL, FFD, DMODEL, 0);
  dim3 gF2(DMODEL / 64, NBL / 64);
  gemm_bf16_kernel<0, false><<<gF2, 128, 0, stream>>>(h1, W2T, b2, tmpF, nullptr, NBL, DMODEL, FFD, 0);

  // 7) out = LN(x + ff)
  add_ln_kernel<<<NBL, 256, 0, stream>>>(tmpF, xF, ln2g, ln2b, (float*)d_out, nullptr);
}
